// HybridBlock_9277129359804
// MI455X (gfx1250) — matrix operand order repaired and checked by reading, not compiled, not run on hardware
//
#include <hip/hip_runtime.h>
#include <hip/hip_bf16.h>

typedef __bf16  bf16_t;
typedef __attribute__((ext_vector_type(16))) __bf16 v16bf;
typedef __attribute__((ext_vector_type(8)))  float  v8f;
typedef int v4i __attribute__((vector_size(16)));

union ABf { v16bf v; unsigned short u[16]; };

__device__ __forceinline__ unsigned short f2bf(float f) {
    unsigned int x = __float_as_uint(f);
    unsigned int r = x + 0x7FFFu + ((x >> 16) & 1u);   // RNE
    return (unsigned short)(r >> 16);
}
__device__ __forceinline__ float bf2f(unsigned short u) {
    return __uint_as_float(((unsigned int)u) << 16);
}
__device__ __forceinline__ v8f wmma_bf(v16bf a, v16bf b, v8f c) {
    return __builtin_amdgcn_wmma_f32_16x16x32_bf16(false, a, false, b, (short)0, c, false, false);
}
__device__ __forceinline__ v8f vzero8() {
    v8f z;
#pragma unroll
    for (int i = 0; i < 8; ++i) z[i] = 0.f;
    return z;
}

// ---------------------------------------------------------------------------
// CDNA5 async global->LDS copy (ASYNCcnt path), with safe fallback.
// Builtin signature is (v4i AS1*, v4i AS3*, imm offset, imm cpol).
// AS(3) pointer = low 32 bits of generic LDS address (ISA §10.2 aperture map).
// ---------------------------------------------------------------------------
#if __has_builtin(__builtin_amdgcn_global_load_async_to_lds_b128) && \
    __has_builtin(__builtin_amdgcn_s_wait_asynccnt)
#define HAS_ASYNC_LDS 1
#else
#define HAS_ASYNC_LDS 0
#endif

__device__ __forceinline__ void async_copy16(void* lds_dst, const void* g_src) {
#if HAS_ASYNC_LDS
    __builtin_amdgcn_global_load_async_to_lds_b128(
        (__attribute__((address_space(1))) v4i*)(unsigned long long)g_src,
        (__attribute__((address_space(3))) v4i*)(unsigned int)(unsigned long long)lds_dst,
        0, 0);
#else
    *(uint4*)lds_dst = *(const uint4*)g_src;
#endif
}
__device__ __forceinline__ void async_wait0() {
#if HAS_ASYNC_LDS
    __builtin_amdgcn_s_wait_asynccnt(0);
#endif
}

// ---------------------------------------------------------------------------
// fp32 -> bf16 conversion
// ---------------------------------------------------------------------------
__global__ void cvt_kernel(const float* __restrict__ s, unsigned short* __restrict__ d, int n) {
    int i = blockIdx.x * 256 + threadIdx.x;
    if (i < n) d[i] = f2bf(s[i]);
}

// ---------------------------------------------------------------------------
// Row LayerNorm (D elems per row), optional f32 and bf16 outputs
// ---------------------------------------------------------------------------
__global__ __launch_bounds__(256) void ln_kernel(
    const float* __restrict__ in, const float* __restrict__ g, const float* __restrict__ b,
    float* __restrict__ outf, unsigned short* __restrict__ outbf, int D)
{
    __shared__ float red[256];
    int row = blockIdx.x, tid = threadIdx.x;
    const float* x = in + (long)row * D;
    float s = 0.f, s2 = 0.f;
    for (int c = tid; c < D; c += 256) { float v = x[c]; s += v; s2 += v * v; }
    red[tid] = s; __syncthreads();
    for (int o = 128; o > 0; o >>= 1) { if (tid < o) red[tid] += red[tid + o]; __syncthreads(); }
    float mean = red[0] / (float)D; __syncthreads();
    red[tid] = s2; __syncthreads();
    for (int o = 128; o > 0; o >>= 1) { if (tid < o) red[tid] += red[tid + o]; __syncthreads(); }
    float var = red[0] / (float)D - mean * mean;
    float inv = rsqrtf(var + 1e-5f);
    for (int c = tid; c < D; c += 256) {
        float v = (x[c] - mean) * inv * g[c] + b[c];
        if (outf)  outf[(long)row * D + c]  = v;
        if (outbf) outbf[(long)row * D + c] = f2bf(v);
    }
}

// ---------------------------------------------------------------------------
// Generic bf16 WMMA GEMM: C(MxN,f32) = A(MxK) @ B(KxN) + bias (+resid) (act)
// Block 256 = 8 waves, each wave 16(M)x128(N), workgroup tile 128x128.
// B tile staged in LDS K-major so B fragments are contiguous 128b ds reads.
// ---------------------------------------------------------------------------
__global__ __launch_bounds__(256) void gemm_bf16_kernel(
    const unsigned short* __restrict__ A, const unsigned short* __restrict__ B,
    const float* __restrict__ bias, const float* __restrict__ resid,
    float* __restrict__ C, unsigned short* __restrict__ Cbf,
    int M, int N, int K, int act)
{
    __shared__ __align__(16) unsigned short bsh[128 * 32];   // [col][k]
    int lane = threadIdx.x & 31, wave = threadIdx.x >> 5;
    int m0 = blockIdx.y * 128 + wave * 16;
    int n0 = blockIdx.x * 128;

    v8f acc[8];
#pragma unroll
    for (int i = 0; i < 8; ++i) acc[i] = vzero8();

    for (int k0 = 0; k0 < K; k0 += 32) {
        // stage B[k0..k0+32) x [n0..n0+128) -> LDS (K-major per column)
        {
            int t   = threadIdx.x;
            int row = t >> 3;            // 0..31 (k within tile)
            int c0  = (t & 7) * 16;      // 0..112
            const unsigned short* src = B + (long)(k0 + row) * N + n0 + c0;
            union { uint4 q[2]; unsigned short u[16]; } tmp;
            tmp.q[0] = *(const uint4*)src;
            tmp.q[1] = *(const uint4*)(src + 8);
            if (k0 + 32 < K) __builtin_prefetch(src + (long)32 * N, 0, 1);
#pragma unroll
            for (int e = 0; e < 16; ++e) bsh[(c0 + e) * 32 + row] = tmp.u[e];
        }
        __syncthreads();

        // A fragment (16x32) — ISA A-layout: lane<16 rows, elems {k..k+7, k+16..k+23}
        ABf af;
        {
            int r  = m0 + (lane & 15);
            int kb = k0 + ((lane >> 4) << 3);
            const unsigned short* p = A + (long)r * K + kb;
            *(uint4*)&af.u[0] = *(const uint4*)p;
            *(uint4*)&af.u[8] = *(const uint4*)(p + 16);
        }
#pragma unroll
        for (int nt = 0; nt < 8; ++nt) {
            ABf bfr;
            int col = nt * 16 + (lane & 15);
            int kb2 = (lane >> 4) * 8;
            *(uint4*)&bfr.u[0] = *(const uint4*)&bsh[col * 32 + kb2];
            *(uint4*)&bfr.u[8] = *(const uint4*)&bsh[col * 32 + kb2 + 16];
            acc[nt] = wmma_bf(af.v, bfr.v, acc[nt]);
        }
        __syncthreads();
    }

    // epilogue: C layout — row = m0 + (lane>=16)*8 + g, col = n0 + nt*16 + lane&15
#pragma unroll
    for (int nt = 0; nt < 8; ++nt) {
        int col = n0 + nt * 16 + (lane & 15);
        float bv = bias ? bias[col] : 0.f;
#pragma unroll
        for (int g = 0; g < 8; ++g) {
            int row = m0 + ((lane >> 4) << 3) + g;
            long idx = (long)row * N + col;
            float v = acc[nt][g] + bv;
            if (resid) v += resid[idx];
            if (act == 1) v = 0.5f * v * (1.0f + erff(v * 0.70710678118654752f));  // exact GELU
            if (C)   C[idx]   = v;
            if (Cbf) Cbf[idx] = f2bf(v);
        }
    }
}

// ---------------------------------------------------------------------------
// Sliding-window attention (flash style). 1 wave per (head, 16-query tile).
// qkv bf16 (L x 2304) laid out [l][s*768 + h*64 + e]. WINDOW=256 (j<=i, j>=i-255).
// V chunk is copied global->LDS with async DMA at the top of each key chunk so
// the copy overlaps the score WMMAs + streaming softmax (ASYNCcnt hides it).
// ---------------------------------------------------------------------------
__global__ __launch_bounds__(32) void win_attn_kernel(
    const unsigned short* __restrict__ qkv, unsigned short* __restrict__ ctx, int L)
{
    __shared__ __align__(16) unsigned short psh[16 * 32];   // P tile (C->A relayout)
    __shared__ __align__(16) unsigned short vsh[32 * 64];   // V chunk, row-major
    const int D3 = 2304;
    int lane = threadIdx.x & 31;
    int q0 = blockIdx.x * 16;
    int h  = blockIdx.y;
    (void)L;

    // Q fragments (16 rows x 64 feat -> 2 A-frags)
    ABf qf[2];
    {
        int r  = q0 + (lane & 15);
        int kb = ((lane >> 4) << 4);
        const unsigned short* p = qkv + (long)r * D3 + h * 64;
#pragma unroll
        for (int t = 0; t < 2; ++t) {
            const unsigned short* pp = p + t * 32 + kb;
            *(uint4*)&qf[t].u[0] = *(const uint4*)pp;
            *(uint4*)&qf[t].u[8] = *(const uint4*)(pp + 8);
        }
    }

    float mrun[8], lrun[8];
    v8f o[4];
#pragma unroll
    for (int g = 0; g < 8; ++g) { mrun[g] = -1e30f; lrun[g] = 0.f; }
#pragma unroll
    for (int i = 0; i < 4; ++i) o[i] = vzero8();

    int js = q0 - 255; if (js < 0) js = 0;
    int kc0 = js & ~31;
    for (int kc = kc0; kc <= q0 + 15; kc += 32) {
        // ---- kick off async V copy: rows [kc, kc+32) x 64 feats, one row/lane
        {
            const unsigned short* vrow = qkv + (long)(kc + lane) * D3 + 1536 + h * 64;
            unsigned short* dst = vsh + lane * 64;
#pragma unroll
            for (int t = 0; t < 8; ++t) async_copy16(dst + t * 8, vrow + t * 8);
        }

        // ---- scores S = Q @ K^T  (16 x 32, two 16x16 tiles; inner E=64 -> 2 WMMA)
        v8f s[2]; s[0] = vzero8(); s[1] = vzero8();
#pragma unroll
        for (int kt = 0; kt < 2; ++kt) {
            int key = kc + kt * 16 + (lane & 15);
            const unsigned short* kb = qkv + (long)key * D3 + 768 + h * 64 + ((lane >> 4) << 4);
            ABf b0, b1;
            *(uint4*)&b0.u[0] = *(const uint4*)(kb);
            *(uint4*)&b0.u[8] = *(const uint4*)(kb + 8);
            *(uint4*)&b1.u[0] = *(const uint4*)(kb + 32);
            *(uint4*)&b1.u[8] = *(const uint4*)(kb + 40);
            s[kt] = wmma_bf(qf[0].v, b0.v, s[kt]);
            s[kt] = wmma_bf(qf[1].v, b1.v, s[kt]);
        }
        // ---- scale + window mask + streaming softmax
        float rmax[8];
#pragma unroll
        for (int g = 0; g < 8; ++g) {
            int i = q0 + ((lane >> 4) << 3) + g;
            float best = -1e30f;
#pragma unroll
            for (int kt = 0; kt < 2; ++kt) {
                int j = kc + kt * 16 + (lane & 15);
                float v = s[kt][g] * 0.125f;
                if (j > i || j < i - 255) v = -1e4f;
                s[kt][g] = v;
                best = fmaxf(best, v);
            }
            rmax[g] = best;
        }
#pragma unroll
        for (int m = 1; m < 16; m <<= 1)
#pragma unroll
            for (int g = 0; g < 8; ++g)
                rmax[g] = fmaxf(rmax[g], __shfl_xor(rmax[g], m, 16));

#pragma unroll
        for (int g = 0; g < 8; ++g) {
            float newm = fmaxf(mrun[g], rmax[g]);
            float corr = __expf(mrun[g] - newm);
            mrun[g] = newm;
            float rsum = 0.f;
#pragma unroll
            for (int kt = 0; kt < 2; ++kt) {
                float p = __expf(s[kt][g] - newm);
                s[kt][g] = p;
                rsum += p;
            }
#pragma unroll
            for (int m = 1; m < 16; m <<= 1) rsum += __shfl_xor(rsum, m, 16);
            lrun[g] = lrun[g] * corr + rsum;
#pragma unroll
            for (int nt = 0; nt < 4; ++nt) o[nt][g] *= corr;
        }
        // ---- C-layout -> A-layout via LDS round trip (intra-wave ds ordering)
#pragma unroll
        for (int kt = 0; kt < 2; ++kt)
#pragma unroll
            for (int g = 0; g < 8; ++g)
                psh[(((lane >> 4) << 3) + g) * 32 + kt * 16 + (lane & 15)] = f2bf(s[kt][g]);
        ABf pf;
        {
            int r  = lane & 15;
            int kb = ((lane >> 4) << 4);
            const unsigned short* pp = psh + r * 32 + kb;
            *(uint4*)&pf.u[0] = *(const uint4*)pp;
            *(uint4*)&pf.u[8] = *(const uint4*)(pp + 8);
        }
        // ---- O += P @ V  (V fragments now read from LDS after async completes)
        async_wait0();
#pragma unroll
        for (int nt = 0; nt < 4; ++nt) {
            ABf vf;
            int col = nt * 16 + (lane & 15);
            const unsigned short* vp = vsh + ((lane >> 4) << 4) * 64 + col;
#pragma unroll
            for (int e = 0; e < 16; ++e) vf.u[e] = vp[e * 64];
            o[nt] = wmma_bf(pf.v, vf.v, o[nt]);
        }
    }
    // write normalized context (bf16) to (L, 768) at col h*64+e
#pragma unroll
    for (int nt = 0; nt < 4; ++nt)
#pragma unroll
        for (int g = 0; g < 8; ++g) {
            int i = q0 + ((lane >> 4) << 3) + g;
            int e = nt * 16 + (lane & 15);
            ctx[(long)i * 768 + h * 64 + e] = f2bf(o[nt][g] / lrun[g]);
        }
}

// ---------------------------------------------------------------------------
// Linear-attention feature map: q=(elu(x)+1+1e-4)*scale, k=(elu(x)+1+1e-4), v
// ---------------------------------------------------------------------------
__global__ void eluq_kernel(const float* __restrict__ qkv2,
                            unsigned short* __restrict__ q2, unsigned short* __restrict__ k2,
                            unsigned short* __restrict__ v2, int n)
{
    int idx = blockIdx.x * 256 + threadIdx.x;
    if (idx >= n) return;
    int l = idx / 768, c = idx % 768;
    const float* row = qkv2 + (long)l * 2304;
    float qv = row[c], kv = row[c + 768], vv = row[c + 1536];
    qv = (qv > 0.f ? qv + (1.0f + 1e-4f) : __expf(qv) + 1e-4f) * 0.125f;
    kv = (kv > 0.f ? kv + (1.0f + 1e-4f) : __expf(kv) + 1e-4f);
    q2[idx] = f2bf(qv); k2[idx] = f2bf(kv); v2[idx] = f2bf(vv);
}

// ---------------------------------------------------------------------------
// Linear attention pass 1: per (head, chunk of 64) S_c = K_c^T V_c (64x64),
// ksum_c[e] = sum_j k.  4 waves, each wave a 16-row strip of S.
// ---------------------------------------------------------------------------
__global__ __launch_bounds__(128) void lin_chunk_kv_kernel(
    const unsigned short* __restrict__ k2, const unsigned short* __restrict__ v2,
    float* __restrict__ S, float* __restrict__ ksum)
{
    const int D = 768;
    int c = blockIdx.x, h = blockIdx.y;
    int lane = threadIdx.x & 31, wave = threadIdx.x >> 5;
    const unsigned short* kb = k2 + (long)c * 64 * D + h * 64;
    const unsigned short* vb = v2 + (long)c * 64 * D + h * 64;

    v8f acc[4];
#pragma unroll
    for (int i = 0; i < 4; ++i) acc[i] = vzero8();

    for (int j0 = 0; j0 < 64; j0 += 32) {
        // A = K^T (E x C): elem (e, j) = kb[j*D + e]
        ABf af;
        {
            int e  = wave * 16 + (lane & 15);
            int jb = j0 + ((lane >> 4) << 4);
#pragma unroll
            for (int t = 0; t < 8; ++t) af.u[t]     = kb[(long)(jb + t) * D + e];
#pragma unroll
            for (int t = 0; t < 8; ++t) af.u[8 + t] = kb[(long)(jb + 8 + t) * D + e];
        }
#pragma unroll
        for (int nt = 0; nt < 4; ++nt) {
            ABf bfr;
            int f = nt * 16 + (lane & 15);
            int jf = j0 + ((lane >> 4) << 4);
#pragma unroll
            for (int e = 0; e < 16; ++e) bfr.u[e] = vb[(long)(jf + e) * D + f];
            acc[nt] = wmma_bf(af.v, bfr.v, acc[nt]);
        }
    }
    float* Sc = S + ((long)h * 32 + c) * 4096;
#pragma unroll
    for (int nt = 0; nt < 4; ++nt)
#pragma unroll
        for (int g = 0; g < 8; ++g) {
            int e = wave * 16 + ((lane >> 4) << 3) + g;
            int f = nt * 16 + (lane & 15);
            Sc[e * 64 + f] = acc[nt][g];
        }
    int t = threadIdx.x;
    if (t < 64) {
        float s = 0.f;
        for (int j = 0; j < 64; ++j) s += bf2f(kb[(long)j * D + t]);
        ksum[((long)h * 32 + c) * 64 + t] = s;
    }
}

// Exclusive prefix scan across the 32 chunks (S: 12*4096 lanes, ksum: 12*64)
__global__ void lin_scan_kernel(float* __restrict__ S, float* __restrict__ ksum)
{
    int idx = blockIdx.x * 256 + threadIdx.x;
    const int NS = 12 * 4096;
    if (idx < NS) {
        int h = idx >> 12, ef = idx & 4095;
        float run = 0.f;
        for (int c = 0; c < 32; ++c) {
            float* p = S + ((long)h * 32 + c) * 4096 + ef;
            float t = *p; *p = run; run += t;
        }
    } else if (idx < NS + 12 * 64) {
        int r = idx - NS; int h = r >> 6, e = r & 63;
        float run = 0.f;
        for (int c = 0; c < 32; ++c) {
            float* p = ksum + ((long)h * 32 + c) * 64 + e;
            float t = *p; *p = run; run += t;
        }
    }
}

// ---------------------------------------------------------------------------
// Linear attention pass 2: out_i = (q_i@Spref + sum_{j<=i} (q_i.k_j) v_j) / den
// den_i = q_i.kpref + rowsum(masked scores) + 1e-4.  4 waves / chunk.
// ---------------------------------------------------------------------------
__global__ __launch_bounds__(128) void lin_chunk_out_kernel(
    const unsigned short* __restrict__ q2, const unsigned short* __restrict__ k2,
    const unsigned short* __restrict__ v2, const float* __restrict__ S,
    const float* __restrict__ ksum, unsigned short* __restrict__ out)
{
    __shared__ __align__(16) unsigned short psh[4][16 * 64];
    const int D = 768;
    int c = blockIdx.x, h = blockIdx.y;
    int lane = threadIdx.x & 31, wave = threadIdx.x >> 5;
    long base = (long)c * 64 * D + h * 64;

    // Q A-frags for this wave's 16 rows
    ABf qf[2];
    {
        int r = c * 64 + wave * 16 + (lane & 15);
        const unsigned short* p = q2 + (long)r * D + h * 64;
        int kb = ((lane >> 4) << 4);
#pragma unroll
        for (int t = 0; t < 2; ++t) {
            *(uint4*)&qf[t].u[0] = *(const uint4*)(p + t * 32 + kb);
            *(uint4*)&qf[t].u[8] = *(const uint4*)(p + t * 32 + kb + 8);
        }
    }
    // scores (16 x 64)
    v8f sc[4];
#pragma unroll
    for (int i = 0; i < 4; ++i) sc[i] = vzero8();
#pragma unroll
    for (int jt = 0; jt < 4; ++jt) {
        int key = c * 64 + jt * 16 + (lane & 15);
        const unsigned short* kp = k2 + (long)key * D + h * 64 + ((lane >> 4) << 4);
        ABf b0, b1;
        *(uint4*)&b0.u[0] = *(const uint4*)(kp);
        *(uint4*)&b0.u[8] = *(const uint4*)(kp + 8);
        *(uint4*)&b1.u[0] = *(const uint4*)(kp + 32);
        *(uint4*)&b1.u[8] = *(const uint4*)(kp + 40);
        sc[jt] = wmma_bf(qf[0].v, b0.v, sc[jt]);
        sc[jt] = wmma_bf(qf[1].v, b1.v, sc[jt]);
    }
    // causal mask (j<=i), rowsum -> den_intra, store masked bf16 strip
    float den[8];
#pragma unroll
    for (int g = 0; g < 8; ++g) {
        int irel = wave * 16 + ((lane >> 4) << 3) + g;
        float rs = 0.f;
#pragma unroll
        for (int jt = 0; jt < 4; ++jt) {
            int jrel = jt * 16 + (lane & 15);
            float v = sc[jt][g];
            if (jrel > irel) v = 0.f;
            sc[jt][g] = v; rs += v;
            psh[wave][(((lane >> 4) << 3) + g) * 64 + jt * 16 + (lane & 15)] = f2bf(v);
        }
#pragma unroll
        for (int m = 1; m < 16; m <<= 1) rs += __shfl_xor(rs, m, 16);
        den[g] = rs;
    }
    // den_inter = q_i . kpref
    const float* kpref = ksum + ((long)h * 32 + c) * 64;
#pragma unroll
    for (int g = 0; g < 8; ++g) {
        int r = c * 64 + wave * 16 + ((lane >> 4) << 3) + g;
        const unsigned short* qp = q2 + (long)r * D + h * 64;
        float d = 0.f;
        for (int e = 0; e < 64; ++e) d += bf2f(qp[e]) * kpref[e];
        den[g] += d + 1e-4f;
    }
    // P A-frags from LDS (own strip only; intra-wave ds ordering)
    ABf pf[2];
    {
        int r = lane & 15, kb = ((lane >> 4) << 4);
#pragma unroll
        for (int t = 0; t < 2; ++t) {
            const unsigned short* pp = &psh[wave][r * 64 + t * 32 + kb];
            *(uint4*)&pf[t].u[0] = *(const uint4*)pp;
            *(uint4*)&pf[t].u[8] = *(const uint4*)(pp + 8);
        }
    }
    const float* Sp = S + ((long)h * 32 + c) * 4096;
    v8f o[4];
#pragma unroll
    for (int i = 0; i < 4; ++i) o[i] = vzero8();
#pragma unroll
    for (int nt = 0; nt < 4; ++nt) {
        int f = nt * 16 + (lane & 15);
#pragma unroll
        for (int kb = 0; kb < 2; ++kb) {
            ABf vf;
            int jf = kb * 32 + ((lane >> 4) << 4);
#pragma unroll
            for (int e = 0; e < 16; ++e) vf.u[e] = v2[base + (long)(jf + e) * D + f];
            o[nt] = wmma_bf(pf[kb].v, vf.v, o[nt]);
            ABf sf;
            int ef = kb * 32 + ((lane >> 4) << 4);
#pragma unroll
            for (int e = 0; e < 16; ++e) sf.u[e] = f2bf(Sp[(ef + e) * 64 + f]);
            o[nt] = wmma_bf(qf[kb].v, sf.v, o[nt]);
        }
    }
#pragma unroll
    for (int nt = 0; nt < 4; ++nt)
#pragma unroll
        for (int g = 0; g < 8; ++g) {
            int irel = wave * 16 + ((lane >> 4) << 3) + g;
            int f = nt * 16 + (lane & 15);
            out[(long)(c * 64 + irel) * D + h * 64 + f] = f2bf(o[nt][g] / den[g]);
        }
}

// combine: bf16(a + b)
__global__ void addbf_kernel(const float* __restrict__ a, const float* __restrict__ b,
                             unsigned short* __restrict__ o, int n)
{
    int i = blockIdx.x * 256 + threadIdx.x;
    if (i < n) o[i] = f2bf(a[i] + b[i]);
}

// ---------------------------------------------------------------------------
extern "C" void kernel_launch(void* const* d_in, const int* in_sizes, int n_in,
                              void* d_out, int out_size, void* d_ws, size_t ws_size,
                              hipStream_t stream)
{
    const int L = 2048, D = 768, H = 12, D3 = 2304, DF = 3072;

    const float* x          = (const float*)d_in[0];
    const float* ln1_g      = (const float*)d_in[1];
    const float* ln1_b      = (const float*)d_in[2];
    const float* attn_qkv_w = (const float*)d_in[3];
    const float* attn_qkv_b = (const float*)d_in[4];
    const float* attn_prj_w = (const float*)d_in[5];
    const float* attn_prj_b = (const float*)d_in[6];
    const float* ssm_qkv_w  = (const float*)d_in[7];
    const float* ssm_qkv_b  = (const float*)d_in[8];
    const float* ssm_out_w  = (const float*)d_in[9];
    const float* ssm_out_b  = (const float*)d_in[10];
    const float* ssm_ln_g   = (const float*)d_in[11];
    const float* ssm_ln_b   = (const float*)d_in[12];
    const float* out_w      = (const float*)d_in[13];
    const float* out_b      = (const float*)d_in[14];
    const float* mlp_w1     = (const float*)d_in[15];
    const float* mlp_b1     = (const float*)d_in[16];
    const float* mlp_w2     = (const float*)d_in[17];
    const float* mlp_b2     = (const float*)d_in[18];
    const float* lnm_g      = (const float*)d_in[19];
    const float* lnm_b      = (const float*)d_in[20];
    float* outp = (float*)d_out;

    char* ws = (char*)d_ws;
    size_t off = 0;
    auto alloc = [&](size_t bytes) -> void* {
        void* p = ws + off;
        off += (bytes + 255) & ~(size_t)255;
        return p;
    };
    typedef unsigned short us;
    us* wqkv_bf   = (us*)alloc((size_t)D * D3 * 2);
    us* wssm_bf   = (us*)alloc((size_t)D * D3 * 2);
    us* wprj_bf   = (us*)alloc((size_t)D * D * 2);
    us* wsso_bf   = (us*)alloc((size_t)D * D * 2);
    us* wout_bf   = (us*)alloc((size_t)D * D * 2);
    us* w1_bf     = (us*)alloc((size_t)D * DF * 2);
    us* w2_bf     = (us*)alloc((size_t)DF * D * 2);
    us* xn_bf     = (us*)alloc((size_t)L * D * 2);
    us* qkv_bf    = (us*)alloc((size_t)L * D3 * 2);
    float* qkv2_f = (float*)alloc((size_t)L * D3 * 4);
    us* q2bf      = (us*)alloc((size_t)L * D * 2);
    us* k2bf      = (us*)alloc((size_t)L * D * 2);
    us* v2bf      = (us*)alloc((size_t)L * D * 2);
    us* ctx_bf    = (us*)alloc((size_t)L * D * 2);
    float* ao_f   = (float*)alloc((size_t)L * D * 4);
    float* Sbuf   = (float*)alloc((size_t)H * 32 * 4096 * 4);
    float* ksumb  = (float*)alloc((size_t)H * 32 * 64 * 4);
    us* lin_bf    = (us*)alloc((size_t)L * D * 2);
    float* sor_f  = (float*)alloc((size_t)L * D * 4);
    float* so_f   = (float*)alloc((size_t)L * D * 4);
    us* comb_bf   = (us*)alloc((size_t)L * D * 2);
    float* x2_f   = (float*)alloc((size_t)L * D * 4);
    us* h_bf      = (us*)alloc((size_t)L * D * 2);
    us* h1_bf     = (us*)alloc((size_t)L * DF * 2);
    (void)ws_size; (void)n_in; (void)in_sizes; (void)out_size;

    auto cvt = [&](const float* s, us* d, int n) {
        cvt_kernel<<<(n + 255) / 256, 256, 0, stream>>>(s, d, n);
    };
    cvt(attn_qkv_w, wqkv_bf, D * D3);
    cvt(ssm_qkv_w,  wssm_bf, D * D3);
    cvt(attn_prj_w, wprj_bf, D * D);
    cvt(ssm_out_w,  wsso_bf, D * D);
    cvt(out_w,      wout_bf, D * D);
    cvt(mlp_w1,     w1_bf,   D * DF);
    cvt(mlp_w2,     w2_bf,   DF * D);

    // xn = LN1(x) -> bf16
    ln_kernel<<<L, 256, 0, stream>>>(x, ln1_g, ln1_b, nullptr, xn_bf, D);

    // qkv = xn @ Wqkv + b (bf16), qkv2 = xn @ Wssm + b (f32)
    gemm_bf16_kernel<<<dim3(D3 / 128, L / 128), 256, 0, stream>>>(
        xn_bf, wqkv_bf, attn_qkv_b, nullptr, nullptr, qkv_bf, L, D3, D, 0);
    gemm_bf16_kernel<<<dim3(D3 / 128, L / 128), 256, 0, stream>>>(
        xn_bf, wssm_bf, ssm_qkv_b, nullptr, qkv2_f, nullptr, L, D3, D, 0);

    // feature maps for linear attention
    eluq_kernel<<<(L * D + 255) / 256, 256, 0, stream>>>(qkv2_f, q2bf, k2bf, v2bf, L * D);

    // sliding-window attention
    win_attn_kernel<<<dim3(L / 16, H), 32, 0, stream>>>(qkv_bf, ctx_bf, L);
    gemm_bf16_kernel<<<dim3(D / 128, L / 128), 256, 0, stream>>>(
        ctx_bf, wprj_bf, attn_prj_b, nullptr, ao_f, nullptr, L, D, D, 0);

    // linear attention (chunked scan)
    lin_chunk_kv_kernel<<<dim3(32, H), 128, 0, stream>>>(k2bf, v2bf, Sbuf, ksumb);
    lin_scan_kernel<<<(H * 4096 + H * 64 + 255) / 256, 256, 0, stream>>>(Sbuf, ksumb);
    lin_chunk_out_kernel<<<dim3(32, H), 128, 0, stream>>>(q2bf, k2bf, v2bf, Sbuf, ksumb, lin_bf);
    gemm_bf16_kernel<<<dim3(D / 128, L / 128), 256, 0, stream>>>(
        lin_bf, wsso_bf, ssm_out_b, nullptr, sor_f, nullptr, L, D, D, 0);
    ln_kernel<<<L, 256, 0, stream>>>(sor_f, ssm_ln_g, ssm_ln_b, so_f, nullptr, D);

    // combine + out projection + residual
    addbf_kernel<<<(L * D + 255) / 256, 256, 0, stream>>>(ao_f, so_f, comb_bf, L * D);
    gemm_bf16_kernel<<<dim3(D / 128, L / 128), 256, 0, stream>>>(
        comb_bf, wout_bf, out_b, x, x2_f, nullptr, L, D, D, 0);

    // MLP
    ln_kernel<<<L, 256, 0, stream>>>(x2_f, lnm_g, lnm_b, nullptr, h_bf, D);
    gemm_bf16_kernel<<<dim3(DF / 128, L / 128), 256, 0, stream>>>(
        h_bf, w1_bf, mlp_b1, nullptr, nullptr, h1_bf, L, DF, D, 1);
    gemm_bf16_kernel<<<dim3(D / 128, L / 128), 256, 0, stream>>>(
        h1_bf, w2_bf, mlp_b2, x2_f, outp, nullptr, L, D, DF, 0);
}
